// Unet_49031346651671
// MI455X (gfx1250) — matrix operand order repaired and checked by reading, not compiled, not run on hardware
//
#include <hip/hip_runtime.h>
#include <string.h>

// ---------------- types for WMMA fragments ----------------
typedef __attribute__((ext_vector_type(16))) __bf16         v16bf;
typedef __attribute__((ext_vector_type(8)))  float          v8f;
typedef __attribute__((ext_vector_type(8)))  unsigned short us8;
typedef __attribute__((ext_vector_type(16))) unsigned short us16;

#define BN_EPS 0.0001f
#define CMAX   448   // max channel count (2*224)

// float -> bf16 (round to nearest even)
__device__ __forceinline__ unsigned short f2bf(float f) {
    unsigned int u = __float_as_uint(f);
    u += 0x7FFFu + ((u >> 16) & 1u);
    return (unsigned short)(u >> 16);
}

// ---------------- BN statistics: per-channel sum / sumsq via atomics ----------------
__global__ void k_stats(const float* __restrict__ x, int n, int stride, int C,
                        int rowsPerBlock, float* __restrict__ sum, float* __restrict__ sq) {
    int r0 = blockIdx.x * rowsPerBlock;
    int r1 = r0 + rowsPerBlock; if (r1 > n) r1 = n;
    for (int c = threadIdx.x; c < C; c += blockDim.x) {
        float s = 0.f, q = 0.f;
        for (int r = r0; r < r1; ++r) {
            float v = x[(size_t)r * stride + c];
            s += v; q += v * v;
        }
        atomicAdd(&sum[c], s);
        atomicAdd(&sq[c], q);
    }
}

// scale = g * rsqrt(var + eps);  bias = b - mean * scale
__global__ void k_finalize(const float* __restrict__ sum, const float* __restrict__ sq,
                           const float* __restrict__ g, const float* __restrict__ b,
                           int C, float invN, float* __restrict__ scale, float* __restrict__ bias) {
    int c = blockIdx.x * blockDim.x + threadIdx.x;
    if (c >= C) return;
    float mean = sum[c] * invN;
    float var  = sq[c] * invN - mean * mean;
    float s = g[c] * rsqrtf(var + BN_EPS);
    scale[c] = s;
    bias[c]  = b[c] - mean * s;
}

// y = act(x*scale+bias) -> bf16 (channel-padded); optional fp32 mirror
__global__ void k_producer(const float* __restrict__ x, int n, int istride,
                           int cReal, int cPad,
                           const float* __restrict__ scale, const float* __restrict__ bias,
                           int relu, unsigned short* __restrict__ xbf,
                           float* __restrict__ mirror) {
    size_t id = (size_t)blockIdx.x * blockDim.x + threadIdx.x;
    size_t tot = (size_t)n * cPad;
    if (id >= tot) return;
    int c = (int)(id % cPad);
    size_t r = id / cPad;
    float v = 0.f;
    if (c < cReal) {
        v = x[r * istride + c];
        if (scale) v = v * scale[c] + bias[c];
        if (relu && v < 0.f) v = 0.f;
    }
    xbf[id] = f2bf(v);
    if (mirror) mirror[id] = v;
}

// W fp32 [taps][ciReal][coReal]  ->  bf16 swizzled [taps][KC][coT][lane(32)][16]
// lane holds column n=lane&15, K = (lane<16 ? 0..7,16..23 : 8..15,24..31) within a 32-chunk
__global__ void k_swizzle(const float* __restrict__ W, int taps, int ciReal, int KC,
                          int coReal, int coT, unsigned short* __restrict__ wswz) {
    size_t id = (size_t)blockIdx.x * blockDim.x + threadIdx.x;
    size_t tot = (size_t)taps * KC * coT * 32 * 16;
    if (id >= tot) return;
    size_t i = id;
    int e      = (int)(i & 15);  i >>= 4;
    int lane   = (int)(i & 31);  i >>= 5;
    int coTile = (int)(i % coT); i /= coT;
    int c      = (int)(i % KC);  i /= KC;
    int k      = (int)i;
    int klocal = ((lane >= 16) ? 8 : 0) + ((e < 8) ? e : 16 + (e - 8));
    int ci = c * 32 + klocal;
    int co = coTile * 16 + (lane & 15);
    float v = 0.f;
    if (ci < ciReal && co < coReal)
        v = W[((size_t)k * ciReal + ci) * coReal + co];
    wswz[id] = f2bf(v);
}

// ---------------- the WMMA gather-GEMM ----------------
// One wave computes a 16-row x 32-col output tile (two 16x16 WMMA tiles sharing
// the gathered A fragment -> half the gather traffic, two independent WMMA chains).
// mode 0: src = g0[row*taps + k]                      (nbr / child gather conv)
// mode 1: src = (g1[row]==k) ? g0[row] : -1           (deconv: parent/poff)
// mode 2: src = row, taps must be 1                   (dense matmul)
#define GW 8   // waves per block
__global__ __launch_bounds__(GW * 32)
void k_gemm(const unsigned short* __restrict__ xbf, int ciPad,
            const unsigned short* __restrict__ wswz,
            const int* __restrict__ g0, const int* __restrict__ g1,
            int mode, int taps, int coReal, int coT,
            int nOut, float* __restrict__ out, int ostride, int ocol0, int accFlag) {
    int lane = threadIdx.x & 31;
    int wave = blockIdx.x * GW + (threadIdx.x >> 5);
    int rowTiles = (nOut + 15) >> 4;
    int coPairs  = (coT + 1) >> 1;
    if (wave >= rowTiles * coPairs) return;
    int rowT   = wave / coPairs;
    int coTile0 = (wave - rowT * coPairs) * 2;
    bool hasT1 = (coTile0 + 1) < coT;
    int rowBase = rowT << 4;
    int myRow = rowBase + (lane & 15);
    bool rowOK = myRow < nOut;
    int kc = ciPad >> 5;
    int off0 = (lane < 16) ? 0 : 8;

    int par = -1, pf = -1;
    if (mode == 1 && rowOK) { par = g0[myRow]; pf = g1[myRow]; }

    v8f acc0 = {};
    v8f acc1 = {};
    for (int k = 0; k < taps; ++k) {
        int src;
        if (mode == 0)      src = rowOK ? g0[(size_t)myRow * taps + k] : -1;
        else if (mode == 1) src = (pf == k) ? par : -1;
        else                src = rowOK ? myRow : -1;

        const unsigned short* xrow = xbf + (size_t)(src < 0 ? 0 : src) * ciPad;
        for (int c = 0; c < kc; ++c) {
            us16 araw;
            if (src >= 0) {
                us8 lo = *(const us8*)(xrow + c * 32 + off0);
                us8 hi = *(const us8*)(xrow + c * 32 + off0 + 16);
                #pragma unroll
                for (int i = 0; i < 8; ++i) { araw[i] = lo[i]; araw[8 + i] = hi[i]; }
            } else {
                #pragma unroll
                for (int i = 0; i < 16; ++i) araw[i] = 0;
            }
            const unsigned short* bbase = wswz +
                ((((size_t)k * kc + c) * coT + coTile0) * 32 + lane) * 16;
            us16 braw0 = *(const us16*)bbase;
            v16bf a  = __builtin_bit_cast(v16bf, araw);
            v16bf b0 = __builtin_bit_cast(v16bf, braw0);
            acc0 = __builtin_amdgcn_wmma_f32_16x16x32_bf16(
                false, a, false, b0, (short)0, acc0, false, false);
            if (hasT1) {
                us16 braw1 = *(const us16*)(bbase + 32 * 16);
                v16bf b1 = __builtin_bit_cast(v16bf, braw1);
                acc1 = __builtin_amdgcn_wmma_f32_16x16x32_bf16(
                    false, a, false, b1, (short)0, acc1, false, false);
            }
        }
    }
    // C/D layout: VGPR v, lane l: M = v + (l<16?0:8), N = l&15
    int n = lane & 15;
    int mBase = (lane < 16) ? 0 : 8;
    #pragma unroll
    for (int t = 0; t < 2; ++t) {
        if (t == 1 && !hasT1) break;
        int col = (coTile0 + t) * 16 + n;
        if (col >= coReal) continue;
        const v8f& acc = (t == 0) ? acc0 : acc1;
        #pragma unroll
        for (int v = 0; v < 8; ++v) {
            int r = rowBase + mBase + v;
            if (r < nOut) {
                size_t o = (size_t)r * ostride + ocol0 + col;
                out[o] = accFlag ? out[o] + acc[v] : acc[v];
            }
        }
    }
}

// copy a column slice (used to build the concat buffer)
__global__ void k_copycols(const float* __restrict__ src, int n, int sstride, int scol0,
                           int C, float* __restrict__ dst, int dstride, int dcol0) {
    size_t id = (size_t)blockIdx.x * blockDim.x + threadIdx.x;
    if (id >= (size_t)n * C) return;
    int c = (int)(id % C);
    size_t r = id / C;
    dst[r * dstride + dcol0 + c] = src[r * sstride + scol0 + c];
}

__global__ void k_biasfill(const float* __restrict__ bias, int n, int C, float* __restrict__ out) {
    size_t id = (size_t)blockIdx.x * blockDim.x + threadIdx.x;
    if (id >= (size_t)n * C) return;
    out[id] = bias[id % C];
}

// feat = leaky(bn(t)); write feat to d_out; h = bf16(feat + xr) for head GEMM
__global__ void k_embed_finish(const float* __restrict__ t,
                               const float* __restrict__ scale, const float* __restrict__ bias,
                               const float* __restrict__ xr, int n, int C,
                               float* __restrict__ featOut, unsigned short* __restrict__ hbf) {
    size_t id = (size_t)blockIdx.x * blockDim.x + threadIdx.x;
    if (id >= (size_t)n * C) return;
    int c = (int)(id % C);
    float f = t[id] * scale[c] + bias[c];
    f = (f > 0.f) ? f : 0.2f * f;
    featOut[id] = f;
    hbf[id] = f2bf(f + xr[id]);
}

// ============================== host ==============================
extern "C" void kernel_launch(void* const* d_in, const int* in_sizes, int n_in,
                              void* d_out, int out_size, void* d_ws, size_t ws_size,
                              hipStream_t stream) {
    (void)n_in; (void)out_size; (void)ws_size;
    int p = 0;
    auto F = [&](int i) { return (const float*)d_in[i]; };

    const float* seg = F(p);
    int N0 = in_sizes[p] / 4; p++;
    const float* stemW = F(p++);
    struct Lp {
        const float *bn1g,*bn1b,*c1,*bn2g,*bn2b,*c2;
        const float *bndg,*bndb,*down,*bnug,*bnub,*up,*nin;
        const float *pbn1g,*pbn1b,*pc1,*pbn2g,*pbn2b,*pc2;
    } LP[7];
    for (int l = 0; l < 7; ++l) {
        LP[l].bn1g=F(p++); LP[l].bn1b=F(p++); LP[l].c1=F(p++);
        LP[l].bn2g=F(p++); LP[l].bn2b=F(p++); LP[l].c2=F(p++);
        if (l < 6) {
            LP[l].bndg=F(p++); LP[l].bndb=F(p++); LP[l].down=F(p++);
            LP[l].bnug=F(p++); LP[l].bnub=F(p++); LP[l].up=F(p++); LP[l].nin=F(p++);
            LP[l].pbn1g=F(p++); LP[l].pbn1b=F(p++); LP[l].pc1=F(p++);
            LP[l].pbn2g=F(p++); LP[l].pbn2b=F(p++); LP[l].pc2=F(p++);
        }
    }
    const float* finalg=F(p++); const float* finalb=F(p++);
    const float* headw =F(p++); const float* headb =F(p++);
    const float* embedw=F(p++); const float* ebng  =F(p++); const float* ebnb=F(p++);

    const int *nbr[7], *child[6], *parent[6], *poff[6];
    int Nl[7];
    for (int l = 0; l < 7; ++l) {
        nbr[l] = (const int*)d_in[p]; Nl[l] = in_sizes[p] / 27; p++;
        if (l < 6) {
            child[l]  = (const int*)d_in[p++];
            parent[l] = (const int*)d_in[p++];
            poff[l]   = (const int*)d_in[p++];
        }
    }

    int A[7]; for (int l = 0; l < 7; ++l) A[l] = 32 * (l + 1);

    // ---- workspace allocator ----
    char* ws = (char*)d_ws; size_t cur = 0;
    auto alloc = [&](size_t bytes) -> void* {
        size_t a = (cur + 255) & ~(size_t)255; cur = a + bytes; return ws + a;
    };
    float* X[7];
    for (int l = 0; l < 7; ++l) X[l] = (float*)alloc((size_t)Nl[l] * A[l] * 4);
    size_t maxNC2 = 0, maxNCa = 0;
    for (int l = 0; l < 7; ++l) {
        size_t s2 = (size_t)Nl[l] * 2 * A[l]; if (s2 > maxNC2) maxNC2 = s2;
        size_t s1 = (size_t)Nl[l] * A[l];     if (s1 > maxNCa) maxNCa = s1;
    }
    float*          xcat = (float*)alloc(maxNC2 * 4);
    float*          bufA = (float*)alloc(maxNCa * 4);
    unsigned short* hbf  = (unsigned short*)alloc(maxNC2 * 2);
    unsigned short* wswz = (unsigned short*)alloc((size_t)27 * CMAX * 224 * 2);
    float*          stats= (float*)alloc(2 * CMAX * 4);      // sum | sq
    float*          scb  = (float*)alloc(2 * CMAX * 4);      // scale | bias
    float*          xr   = (float*)alloc((size_t)N0 * 32 * 4);

    auto runStats = [&](const float* xp, int n, int stride, int C) {
        hipMemsetAsync(stats, 0, 2 * CMAX * sizeof(float), stream);
        int rpb = 512;
        int nb = (n + rpb - 1) / rpb;
        k_stats<<<nb, 256, 0, stream>>>(xp, n, stride, C, rpb, stats, stats + CMAX);
    };
    auto runFin = [&](const float* g, const float* b, int C, int n) {
        k_finalize<<<(C + 255) / 256, 256, 0, stream>>>(stats, stats + CMAX, g, b, C,
                                                        1.0f / (float)n, scb, scb + CMAX);
    };
    auto runProd = [&](const float* xp, int n, int istride, int cReal, int cPad,
                       bool useBN, int relu, float* mirror) {
        size_t tot = (size_t)n * cPad;
        k_producer<<<(int)((tot + 255) / 256), 256, 0, stream>>>(
            xp, n, istride, cReal, cPad, useBN ? scb : nullptr,
            useBN ? scb + CMAX : nullptr, relu, hbf, mirror);
    };
    auto runSwz = [&](const float* W, int taps, int ciReal, int coReal) {
        int ciPad = (ciReal + 31) & ~31;
        int KC = ciPad >> 5, coT = (coReal + 15) / 16;
        size_t tot = (size_t)taps * KC * coT * 32 * 16;
        k_swizzle<<<(int)((tot + 255) / 256), 256, 0, stream>>>(
            W, taps, ciReal, KC, coReal, coT, wswz);
    };
    auto runGemm = [&](int ciPad, int taps, int coReal,
                       const int* g0, const int* g1, int mode,
                       int nOut, float* out, int ostride, int ocol0, int acc) {
        int coT = (coReal + 15) / 16;
        int coPairs = (coT + 1) / 2;
        long tiles = (long)((nOut + 15) / 16) * coPairs;
        int blocks = (int)((tiles + GW - 1) / GW);
        k_gemm<<<blocks, GW * 32, 0, stream>>>(hbf, ciPad, wswz, g0, g1, mode, taps,
                                               coReal, coT, nOut, out, ostride, ocol0, acc);
    };

    // ---- stem: pad seg [N0,4] -> bf16 [N0,32], 27-tap conv -> X[0] ----
    runProd(seg, N0, 4, 4, 32, false, 0, nullptr);
    runSwz(stemW, 27, 4, 32);
    runGemm(32, 27, 32, nbr[0], nullptr, 0, N0, X[0], 32, 0, 0);

    // ---- down pass ----
    for (int l = 0; l < 7; ++l) {
        int a = A[l], n = Nl[l];
        // pre resblock: X[l] += c2(bnrelu(c1(bnrelu(X[l]))))
        runStats(X[l], n, a, a);  runFin(LP[l].bn1g, LP[l].bn1b, a, n);
        runProd(X[l], n, a, a, a, true, 1, nullptr);
        runSwz(LP[l].c1, 27, a, a);
        runGemm(a, 27, a, nbr[l], nullptr, 0, n, bufA, a, 0, 0);
        runStats(bufA, n, a, a);  runFin(LP[l].bn2g, LP[l].bn2b, a, n);
        runProd(bufA, n, a, a, a, true, 1, nullptr);
        runSwz(LP[l].c2, 27, a, a);
        runGemm(a, 27, a, nbr[l], nullptr, 0, n, X[l], a, 0, 1);
        if (l < 6) {
            int b = A[l + 1];
            runStats(X[l], n, a, a);  runFin(LP[l].bndg, LP[l].bndb, a, n);
            runProd(X[l], n, a, a, a, true, 1, nullptr);
            runSwz(LP[l].down, 8, a, b);
            runGemm(a, 8, b, child[l], nullptr, 0, Nl[l + 1], X[l + 1], b, 0, 0);
        }
    }

    // ---- up pass ----
    for (int l = 5; l >= 0; --l) {
        int a = A[l], b = A[l + 1], n = Nl[l], n1 = Nl[l + 1];
        // deconv(bnrelu(bnu, X[l+1])) -> xcat[:, a:2a]
        runStats(X[l + 1], n1, b, b);  runFin(LP[l].bnug, LP[l].bnub, b, n1);
        runProd(X[l + 1], n1, b, b, b, true, 1, nullptr);
        runSwz(LP[l].up, 8, b, a);
        runGemm(b, 8, a, parent[l], poff[l], 1, n, xcat, 2 * a, a, 0);
        // skip copy: X[l] -> xcat[:, 0:a]
        {
            size_t tot = (size_t)n * a;
            k_copycols<<<(int)((tot + 255) / 256), 256, 0, stream>>>(
                X[l], n, a, 0, a, xcat, 2 * a, 0);
        }
        // shortcut: X[l] = xcat @ nin_w
        runProd(xcat, n, 2 * a, 2 * a, 2 * a, false, 0, nullptr);
        runSwz(LP[l].nin, 1, 2 * a, a);
        runGemm(2 * a, 1, a, nullptr, nullptr, 2, n, X[l], a, 0, 0);
        // post c1
        runStats(xcat, n, 2 * a, 2 * a);  runFin(LP[l].pbn1g, LP[l].pbn1b, 2 * a, n);
        runProd(xcat, n, 2 * a, 2 * a, 2 * a, true, 1, nullptr);
        runSwz(LP[l].pc1, 27, 2 * a, a);
        runGemm(2 * a, 27, a, nbr[l], nullptr, 0, n, bufA, a, 0, 0);
        // post c2 (accumulate onto shortcut)
        runStats(bufA, n, a, a);  runFin(LP[l].pbn2g, LP[l].pbn2b, a, n);
        runProd(bufA, n, a, a, a, true, 1, nullptr);
        runSwz(LP[l].pc2, 27, a, a);
        runGemm(a, 27, a, nbr[l], nullptr, 0, n, X[l], a, 0, 1);
    }

    // ---- head: final_bn+relu, embed, embed_bn, leaky, residual, logits ----
    float* logits  = (float*)d_out;
    float* featOut = (float*)d_out + (size_t)N0 * 20;

    runStats(X[0], N0, 32, 32);  runFin(finalg, finalb, 32, N0);
    runProd(X[0], N0, 32, 32, 32, true, 1, xr);   // hbf = bf16(relu'd), xr = fp32 mirror

    runSwz(embedw, 1, 32, 32);
    runGemm(32, 1, 32, nullptr, nullptr, 2, N0, bufA, 32, 0, 0);

    runStats(bufA, N0, 32, 32);  runFin(ebng, ebnb, 32, N0);
    {
        size_t tot = (size_t)N0 * 32;
        k_embed_finish<<<(int)((tot + 255) / 256), 256, 0, stream>>>(
            bufA, scb, scb + CMAX, xr, N0, 32, featOut, hbf);
    }
    {
        size_t tot = (size_t)N0 * 20;
        k_biasfill<<<(int)((tot + 255) / 256), 256, 0, stream>>>(headb, N0, 20, logits);
    }
    runSwz(headw, 1, 32, 20);
    runGemm(32, 1, 20, nullptr, nullptr, 2, N0, logits, 20, 0, 1);
}
